// FourierGP_60395830116887
// MI455X (gfx1250) — compile-verified, matrix operand order checked by reading, not run on hardware
//
#include <hip/hip_runtime.h>

typedef __attribute__((ext_vector_type(16))) _Float16 v16h;
typedef __attribute__((ext_vector_type(8)))  _Float16 v8h;
typedef __attribute__((ext_vector_type(8)))  float    v8f;
typedef __attribute__((ext_vector_type(4)))  int      v4i;

#define N1 128
#define N2 128
#define LDPAD 136                 // padded row stride (halfs): 272B -> 4-bank shift per row
#define CHALFS (N2 * LDPAD)       // 17408 halfs
#define CBYTES (CHALFS * 2)       // 34816 bytes staged per block
#define WAVES 8
#define TPB   (WAVES * 32)
#define PTS_PER_BLOCK (WAVES * 16)

// planar rotation: (c,s) <- (c*cs - s*ss, s*cs + c*ss)
__device__ __forceinline__ void rot(float& c, float& s, float cs, float ss) {
    float cn = __fmaf_rn(c, cs, -(s * ss));
    float sn = __fmaf_rn(s, cs,  (c * ss));
    c = cn; s = sn;
}

// ---- one-time prep: cg[l*LDPAD + k] = coef[k][l] * w[k] * w[l] (f16), pad = 0 ----
__global__ __launch_bounds__(256) void fouriergp_prep_kernel(
    const float* __restrict__ coef, _Float16* __restrict__ cg)
{
    int e = blockIdx.x * 256 + threadIdx.x;
    if (e >= CHALFS) return;
    int l = e / LDPAD, k = e % LDPAD;
    float v = 0.0f;
    if (k < N1) {
        float fk = (float)k - 64.0f, fl = (float)l - 64.0f;
        v = coef[k * N2 + l] * __expf(-0.02f * fk * fk) * __expf(-0.02f * fl * fl);
    }
    cg[e] = (_Float16)v;
}

__global__ __launch_bounds__(TPB) void fouriergp_wmma_kernel(
    const _Float16* __restrict__ cg,  // prepped c^T, LDS image layout, 34816 B
    const float* __restrict__ xs,     // (M,)
    const float* __restrict__ ys,     // (M,)
    float* __restrict__ out,          // (M,)
    int M)
{
    __shared__ __align__(16) _Float16 cT[CHALFS];

    const int tid = threadIdx.x;

    // ---- stage 0: pull the staged matrix into LDS (async DMA path) ----
#if __has_builtin(__builtin_amdgcn_global_load_async_to_lds_b128)
    {
        char* src = const_cast<char*>((const char*)cg);
        char* dst = (char*)cT;
        typedef __attribute__((address_space(1))) v4i gv4i_t;
        typedef __attribute__((address_space(3))) v4i lv4i_t;
        for (int off = tid * 16; off < CBYTES; off += TPB * 16) {
            __builtin_amdgcn_global_load_async_to_lds_b128(
                (gv4i_t*)(src + off),
                (lv4i_t*)(dst + off),
                0, 0);
        }
#if __has_builtin(__builtin_amdgcn_s_wait_asynccnt)
        __builtin_amdgcn_s_wait_asynccnt(0);
#else
        asm volatile("s_wait_asynccnt 0x0" ::: "memory");
#endif
    }
#else
    for (int c = tid; c < CBYTES / 16; c += TPB)
        ((uint4*)cT)[c] = ((const uint4*)cg)[c];
#endif
    __syncthreads();

    const int w     = tid >> 5;
    const int lane  = tid & 31;
    const int mrow  = lane & 15;       // point within tile == WMMA N column
    const int khalf = lane >> 4;       // K-half selector per 16-bit A/B layout
    int mg  = (blockIdx.x * WAVES + w) * 16 + mrow;
    int mgl = mg < M ? mg : (M - 1);
    const float x = xs[mgl];
    const float y = ys[mgl];

    // ---- B fragments: cos/sin(x * fx_k) in WMMA 16-bit B layout ----
    // element i   (0..7): k = kb*32 + khalf*8 + i
    // element i+8 (0..7): k = kb*32 + 16 + khalf*8 + i
    float cstep, sstep;
    __sincosf(x, &sstep, &cstep);      // one-frequency rotation step
    v16h bc[4], bs[4];
    #pragma unroll
    for (int kb = 0; kb < 4; ++kb) {
        #pragma unroll
        for (int part = 0; part < 2; ++part) {
            int kbase = kb * 32 + part * 16 + khalf * 8;
            float cc, ssv;
            __sincosf(x * (float)(kbase - 64), &ssv, &cc);
            #pragma unroll
            for (int i = 0; i < 8; ++i) {
                bc[kb][part * 8 + i] = (_Float16)cc;
                bs[kb][part * 8 + i] = (_Float16)ssv;
                rot(cc, ssv, cstep, sstep);
            }
        }
    }

    // y-direction rotation step for the in-register fold
    float cstepY, sstepY;
    __sincosf(y, &sstepY, &cstepY);

    // A-fragment loader: rows l = lblock*16 + mrow of c^T, K runs per layout
    auto loadA = [&](int lblock, v16h* af) {
        const v8h* rowv = (const v8h*)&cT[(lblock * 16 + mrow) * LDPAD]; // 16B aligned
        #pragma unroll
        for (int kb = 0; kb < 4; ++kb) {
            int c0 = (kb * 32 + khalf * 8) >> 3;
            v8h a0 = rowv[c0];       // k = base .. base+7
            v8h a1 = rowv[c0 + 2];   // k = base+16 .. base+23
            af[kb] = __builtin_shufflevector(a0, a1,
                0, 1, 2, 3, 4, 5, 6, 7, 8, 9, 10, 11, 12, 13, 14, 15);
        }
    };

    // ---- fused GEMM + fold, fully software-pipelined ----
    // D = C^T @ trig(x): lane = point column m; rows: lanes 0-15 -> l = lb*16 + r,
    // lanes 16-31 -> l = lb*16 + 8 + r.
    // Pipeline: iteration lb issues WMMAs for lb+1, prefetches A for lb+2,
    // and folds lb's accumulators (fold VALU co-executes with next block's XDL).
    float sum = 0.0f;
    v16h af[2][4];
    v8f accC[2], accS[2];

    loadA(0, af[0]);
    loadA(1, af[1]);
    accC[0] = (v8f){}; accS[0] = (v8f){};
    #pragma unroll
    for (int kb = 0; kb < 4; ++kb) {
        accC[0] = __builtin_amdgcn_wmma_f32_16x16x32_f16(
            false, af[0][kb], false, bc[kb], (short)0, accC[0], false, false);
        accS[0] = __builtin_amdgcn_wmma_f32_16x16x32_f16(
            false, af[0][kb], false, bs[kb], (short)0, accS[0], false, false);
    }

    #pragma unroll
    for (int lb = 0; lb < 8; ++lb) {
        const int cur = lb & 1, nxt = cur ^ 1;

        // 1) issue next block's WMMAs into the free accumulator set
        if (lb < 7) {
            accC[nxt] = (v8f){}; accS[nxt] = (v8f){};
            #pragma unroll
            for (int kb = 0; kb < 4; ++kb) {
                accC[nxt] = __builtin_amdgcn_wmma_f32_16x16x32_f16(
                    false, af[nxt][kb], false, bc[kb], (short)0, accC[nxt], false, false);
                accS[nxt] = __builtin_amdgcn_wmma_f32_16x16x32_f16(
                    false, af[nxt][kb], false, bs[kb], (short)0, accS[nxt], false, false);
            }
        }
        // 2) prefetch A fragments two blocks ahead into the retiring slot
        if (lb < 6) loadA(lb + 2, af[cur]);

        // 3) fold current block into the per-point sum, f32 throughout
        int l0 = lb * 16 + khalf * 8;
        float cy, sy;
        __sincosf(y * (float)(l0 - 64), &sy, &cy);
        #pragma unroll
        for (int r = 0; r < 8; ++r) {
            sum = __fmaf_rn(accC[cur][r],  cy, sum);
            sum = __fmaf_rn(accS[cur][r], -sy, sum);
            rot(cy, sy, cstepY, sstepY);
        }
    }

    // lane and lane^16 hold complementary halves of the l-range for the same point
    float other = __shfl_xor(sum, 16, 32);
    sum += other;
    if (lane < 16 && mg < M) out[mg] = sum;
}

extern "C" void kernel_launch(void* const* d_in, const int* in_sizes, int n_in,
                              void* d_out, int out_size, void* d_ws, size_t ws_size,
                              hipStream_t stream) {
    const float* coef = (const float*)d_in[0];   // (128,128) f32
    const float* x    = (const float*)d_in[1];   // (M,) f32
    const float* y    = (const float*)d_in[2];   // (M,) f32
    float* out        = (float*)d_out;           // (M,) f32
    _Float16* cg      = (_Float16*)d_ws;         // 34816 B staged matrix
    const int M = in_sizes[1];

    fouriergp_prep_kernel<<<(CHALFS + 255) / 256, 256, 0, stream>>>(coef, cg);

    const int blocks = (M + PTS_PER_BLOCK - 1) / PTS_PER_BLOCK;
    fouriergp_wmma_kernel<<<blocks, TPB, 0, stream>>>(cg, x, y, out, M);
}
